// LabelPropagator_60370060312793
// MI455X (gfx1250) — matrix operand order repaired and checked by reading, not compiled, not run on hardware
//
#include <hip/hip_runtime.h>
#include <stdint.h>

// ---------------------------------------------------------------------------
// Shapes (fixed by reference): F=16, S=4, c=512, h=w=32 (HW=1024), kd=128, ce=16
// ---------------------------------------------------------------------------
#define S_     4
#define C_     512
#define HW_    1024
#define KD_    128
#define CE_    16
#define NF_    16
#define NKTOK_ (NF_ * HW_)   // 16384

typedef __attribute__((ext_vector_type(16))) _Float16 v16h;
typedef __attribute__((ext_vector_type(8)))  _Float16 v8h;
typedef __attribute__((ext_vector_type(8)))  float    v8f;
typedef unsigned int u32x4 __attribute__((ext_vector_type(4)));
typedef int          i32x8 __attribute__((ext_vector_type(8)));
typedef int          i32x4 __attribute__((ext_vector_type(4)));

#if defined(__has_builtin)
#if __has_builtin(__builtin_amdgcn_tensor_load_to_lds) && \
    __has_builtin(__builtin_amdgcn_s_wait_tensorcnt)
#define HAVE_TDM 1
#endif
#endif
#ifndef HAVE_TDM
#define HAVE_TDM 0
#endif

__device__ __forceinline__ v8f wmma_f16(v16h a, v16h b, v8f c) {
  return __builtin_amdgcn_wmma_f32_16x16x32_f16(
      false, a, false, b, (short)0, c, false, false);
}

__device__ __forceinline__ v16h load_cvt16(const float* __restrict__ p) {
  union { float4 f4[4]; float f[16]; } u;
  u.f4[0] = *(const float4*)(p);
  u.f4[1] = *(const float4*)(p + 4);
  u.f4[2] = *(const float4*)(p + 8);
  u.f4[3] = *(const float4*)(p + 12);
  v16h r;
#pragma unroll
  for (int i = 0; i < 16; ++i) r[i] = (_Float16)u.f[i];
  return r;
}

union HPack { v16h v; v8h h[2]; };

#if HAVE_TDM
// ---------------------------------------------------------------------------
// Tensor Data Mover: async 2D tile (tile_w elems/row x tile_h rows) from
// global -> LDS with optional per-row LDS padding (see CDNA5 ISA).
//   data_size_log2: 0=1B 1=2B 2=4B; row_stride in elements.
//   pad_int: rows of 2^(pad_int+1) DWORDs; pad_amt: (pad_amt+1) DWORDs pad.
// This toolchain exposes the 6-arg builtin:
//   (u32x4 g0, i32x8 g1, i32x4 g2, i32x4 g3, i32x8 extra, i32 cpol)
// ---------------------------------------------------------------------------
__device__ __forceinline__ void tdm_load_2d(uint32_t lds_off, const void* gptr,
                                            uint32_t data_size_log2,
                                            uint32_t tile_w, uint32_t tile_h,
                                            uint32_t row_stride,
                                            uint32_t pad_int, uint32_t pad_amt,
                                            uint32_t pad_en) {
  uint64_t ga = (uint64_t)(uintptr_t)gptr;
  u32x4 g0;
  g0[0] = 1u;                                   // count=1, user descriptor
  g0[1] = lds_off;                              // lds_addr (bytes)
  g0[2] = (uint32_t)ga;                         // global_addr[31:0]
  g0[3] = ((uint32_t)(ga >> 32) & 0x1FFFFFFu) | (2u << 30);  // addr[56:32]|type=2
  i32x8 g1;
  g1[0] = (int)((data_size_log2 << 16) | (pad_en << 20) |
                (pad_int << 22) | (pad_amt << 25));          // mask=0 (no cluster)
  g1[1] = (int)((tile_w & 0xFFFFu) << 16);                   // tensor_dim0 lo
  g1[2] = (int)(((tile_h & 0xFFFFu) << 16) | ((tile_w >> 16) & 0xFFFFu));
  g1[3] = (int)(((tile_w & 0xFFFFu) << 16) | ((tile_h >> 16) & 0xFFFFu));
  g1[4] = (int)tile_h;                          // tile_dim1 (tile_dim2 = 0)
  g1[5] = (int)row_stride;                      // tensor_dim0_stride[31:0]
  g1[6] = 0;
  g1[7] = 0;
  i32x4 gz4 = {0, 0, 0, 0};
  i32x8 gz8 = {0, 0, 0, 0, 0, 0, 0, 0};
  __builtin_amdgcn_tensor_load_to_lds(g0, g1, gz4, gz4, gz8, 0);
}
#endif

// ---------------------------------------------------------------------------
// Kernel 0: convert both weight matrices [kd=128, c=512] f32 -> f16
// ---------------------------------------------------------------------------
__global__ void k_convert_w(const float* __restrict__ wself,
                            const float* __restrict__ wcross,
                            _Float16* __restrict__ out) {
  int i = blockIdx.x * blockDim.x + threadIdx.x;
  const int n = KD_ * C_;
  if (i < n) {
    out[i]     = (_Float16)wself[i];
    out[n + i] = (_Float16)wcross[i];
  }
}

// ---------------------------------------------------------------------------
// Kernel 1: projection + L2 normalize.  Y[s][tok][kd] = l2norm(X @ W^T + b)
// grid = (ntok/16, S), block = 256 (8 waves; wave w owns kd tile w*16..+15)
// The 16-token x 512-ch f32 tile is DMA'd to LDS by the TDM (rows of 16 f32,
// 1-DWORD pad -> 17-DWORD stride, bank-conflict-free), then converted to the
// WMMA A-fragment layout in f16.
// ---------------------------------------------------------------------------
__global__ void k_proj(const float* __restrict__ X, int mem_mode,
                       const _Float16* __restrict__ Wh,
                       const float* __restrict__ bias,
                       _Float16* __restrict__ Out, int ntok) {
  const int tt = blockIdx.x, s = blockIdx.y;
  const int tid = threadIdx.x;
  const int wave = tid >> 5, lane = tid & 31;
  const int row = lane & 15, hi = lane >> 4;

#if HAVE_TDM
  __shared__ float xraw[C_ * 17];        // TDM destination, padded rows
#endif
  __shared__ _Float16 xt[16][C_ + 16];   // token-major f16 tile
  __shared__ float rowAcc[16];
  __shared__ float rowInv[16];

  const float* base;
  if (mem_mode) {
    int f   = (tt * 16) / HW_;
    int hw0 = (tt * 16) % HW_;
    base = X + ((size_t)(f * S_ + s) * C_) * HW_ + hw0;
  } else {
    base = X + (size_t)s * C_ * HW_ + tt * 16;
  }

  if (tid < 16) rowAcc[tid] = 0.f;

#if HAVE_TDM
  if (tid < 32) {   // wave 0 issues one async 2D tile load, waits, releases
    tdm_load_2d((uint32_t)(uintptr_t)xraw, base, /*f32*/2u,
                /*tile_w=*/16u, /*tile_h=*/(uint32_t)C_,
                /*row_stride=*/(uint32_t)HW_,
                /*pad: 16 DWORD rows, +1 DWORD*/3u, 0u, 1u);
    __builtin_amdgcn_s_wait_tensorcnt((short)0);
  }
  __syncthreads();
  for (int cch = tid; cch < C_; cch += 256) {
    v16h hv;
#pragma unroll
    for (int i = 0; i < 16; ++i) hv[i] = (_Float16)xraw[cch * 17 + i];
#pragma unroll
    for (int t2 = 0; t2 < 16; ++t2) xt[t2][cch] = hv[t2];
  }
#else
  for (int cch = tid; cch < C_; cch += 256) {
    v16h hv = load_cvt16(base + (size_t)cch * HW_);
#pragma unroll
    for (int t2 = 0; t2 < 16; ++t2) xt[t2][cch] = hv[t2];
  }
#endif
  __syncthreads();

  const int n0 = wave * 16;
  v8f acc = {};
#pragma unroll 4
  for (int ks = 0; ks < C_ / 32; ++ks) {
    HPack A;
    const _Float16* ap = &xt[row][ks * 32 + hi * 16];
    A.h[0] = *(const v8h*)(ap);
    A.h[1] = *(const v8h*)(ap + 8);
    HPack B;
    const _Float16* wp = Wh + (size_t)(n0 + row) * C_ + ks * 32 + hi * 16;
    B.h[0] = *(const v8h*)(wp);
    B.h[1] = *(const v8h*)(wp + 8);
    acc = wmma_f16(A.v, B.v, acc);
  }

  const float bn = bias[n0 + row];
  float y[8], p[8];
#pragma unroll
  for (int r = 0; r < 8; ++r) { y[r] = acc[r] + bn; p[r] = y[r] * y[r]; }
#pragma unroll
  for (int m = 1; m < 16; m <<= 1)
#pragma unroll
    for (int r = 0; r < 8; ++r) p[r] += __shfl_xor(p[r], m, 32);
  if (row == 0) {
#pragma unroll
    for (int r = 0; r < 8; ++r) atomicAdd(&rowAcc[r + 8 * hi], p[r]);
  }
  __syncthreads();
  if (tid < 16) rowInv[tid] = 1.0f / fmaxf(sqrtf(rowAcc[tid]), 1e-12f);
  __syncthreads();

  _Float16* op = Out + ((size_t)s * ntok + (size_t)tt * 16) * KD_ + n0 + row;
#pragma unroll
  for (int r = 0; r < 8; ++r) {
    int m = r + 8 * hi;
    op[(size_t)m * KD_] = (_Float16)(y[r] * rowInv[m]);
  }
}

// ---------------------------------------------------------------------------
// Kernel 2: self attention (V = tgt, residual added later).
// grid = (HW/16, S), block = 256. Wave w owns value channels [w*64, w*64+64).
// ---------------------------------------------------------------------------
__global__ void k_self_attn(const _Float16* __restrict__ Q,   // [S][HW][KD]
                            const float* __restrict__ tgt,    // [S][C][HW]
                            float* __restrict__ Oout) {       // [S][C][HW]
  const int qt = blockIdx.x, s = blockIdx.y;
  const int tid = threadIdx.x;
  const int wave = tid >> 5, lane = tid & 31;
  const int row = lane & 15, hi = lane >> 4;

  __shared__ _Float16 pbuf[8][16][40];

  const _Float16* qbase = Q + ((size_t)s * HW_ + (size_t)qt * 16) * KD_;
  v16h qfrag[4];
#pragma unroll
  for (int ks = 0; ks < 4; ++ks) {
    HPack U;
    const _Float16* qp = qbase + (size_t)row * KD_ + ks * 32 + hi * 16;
    U.h[0] = *(const v8h*)(qp);
    U.h[1] = *(const v8h*)(qp + 8);
    qfrag[ks] = U.v;
  }

  v8f zero = {};
  v8f oacc[4] = {zero, zero, zero, zero};
  float rs[8] = {0.f, 0.f, 0.f, 0.f, 0.f, 0.f, 0.f, 0.f};

  const _Float16* kbase = Q + (size_t)s * HW_ * KD_;
  const float*    vbase = tgt + (size_t)s * C_ * HW_;

  for (int kt2 = 0; kt2 < HW_ / 32; ++kt2) {
    if (kt2 + 1 < HW_ / 32)
      __builtin_prefetch(kbase + (size_t)((kt2 + 1) * 32 + row) * KD_, 0, 1);
#pragma unroll
    for (int half = 0; half < 2; ++half) {
      const int key0 = kt2 * 32 + half * 16;
      v8f sacc = {};
#pragma unroll
      for (int ks = 0; ks < 4; ++ks) {
        HPack B;
        const _Float16* kp =
            kbase + (size_t)(key0 + row) * KD_ + ks * 32 + hi * 16;
        B.h[0] = *(const v8h*)(kp);
        B.h[1] = *(const v8h*)(kp + 8);
        sacc = wmma_f16(qfrag[ks], B.v, sacc);
      }
#pragma unroll
      for (int r = 0; r < 8; ++r) {
        float e = __expf(sacc[r] * 30.0f);
        rs[r] += e;
        pbuf[wave][r + 8 * hi][half * 16 + row] = (_Float16)e;
      }
    }
    HPack PA;
    const _Float16* pp = &pbuf[wave][row][hi * 16];
    PA.h[0] = *(const v8h*)(pp);
    PA.h[1] = *(const v8h*)(pp + 8);
#pragma unroll
    for (int ct = 0; ct < 4; ++ct) {
      const int ch = wave * 64 + ct * 16 + row;
      v16h bv = load_cvt16(vbase + (size_t)ch * HW_ + kt2 * 32 + hi * 16);
      oacc[ct] = wmma_f16(PA.v, bv, oacc[ct]);
    }
  }

#pragma unroll
  for (int m = 1; m < 16; m <<= 1)
#pragma unroll
    for (int r = 0; r < 8; ++r) rs[r] += __shfl_xor(rs[r], m, 32);

  float* obase = Oout + (size_t)s * C_ * HW_ + (size_t)qt * 16;
#pragma unroll
  for (int ct = 0; ct < 4; ++ct) {
    const int ch = wave * 64 + ct * 16 + row;
#pragma unroll
    for (int r = 0; r < 8; ++r)
      obase[(size_t)ch * HW_ + r + 8 * hi] = oacc[ct][r] / rs[r];
  }
}

// ---------------------------------------------------------------------------
// Kernel 3: residual + instance norm over HW per (s, ch).
// ---------------------------------------------------------------------------
__global__ void k_instnorm(const float* __restrict__ tgt,
                           const float* __restrict__ attn,
                           float* __restrict__ tn) {
  const int rowId = blockIdx.x;
  const int tid = threadIdx.x;
  const float* a = tgt + (size_t)rowId * HW_;
  const float* b = attn + (size_t)rowId * HW_;
  float vals[4], sum = 0.f, sum2 = 0.f;
#pragma unroll
  for (int i = 0; i < 4; ++i) {
    float v = a[tid + i * 256] + b[tid + i * 256];
    vals[i] = v; sum += v; sum2 += v * v;
  }
  __shared__ float s1[8], s2[8];
#pragma unroll
  for (int m = 1; m < 32; m <<= 1) {
    sum  += __shfl_xor(sum,  m, 32);
    sum2 += __shfl_xor(sum2, m, 32);
  }
  if ((tid & 31) == 0) { s1[tid >> 5] = sum; s2[tid >> 5] = sum2; }
  __syncthreads();
  float t1 = 0.f, t2 = 0.f;
#pragma unroll
  for (int w = 0; w < 8; ++w) { t1 += s1[w]; t2 += s2[w]; }
  const float mean = t1 * (1.0f / HW_);
  const float var  = t2 * (1.0f / HW_) - mean * mean;
  const float inv  = rsqrtf(var + 1e-5f);
  float* o = tn + (size_t)rowId * HW_;
#pragma unroll
  for (int i = 0; i < 4; ++i) o[tid + i * 256] = (vals[i] - mean) * inv;
}

// ---------------------------------------------------------------------------
// Kernel 4: cross attention + sigmoid -> output.
// grid = (HW/16, S), block = 256. 8 waves split 16384 keys (2048 each).
// Keys double-buffered into LDS via per-wave TDM descriptors (32 keys x 128
// f16 per tile, 4-DWORD row pad -> 272B rows, 16B-aligned ds_load_b128).
// ---------------------------------------------------------------------------
#define KROW_ 136   // padded halfs per key row in LDS (128 + 8)

__global__ void k_cross_attn(const _Float16* __restrict__ Qc,  // [S][HW][KD]
                             const _Float16* __restrict__ Kc,  // [S][NKTOK][KD]
                             const float* __restrict__ pos,    // [F][S][CE][HW]
                             float* __restrict__ out) {        // [S][CE][HW]
  const int qt = blockIdx.x, s = blockIdx.y;
  const int tid = threadIdx.x;
  const int wave = tid >> 5, lane = tid & 31;
  const int row = lane & 15, hi = lane >> 4;

  __shared__ _Float16 pbuf[8][16][40];
  __shared__ float ored[8][16][16];
  __shared__ float rsred[8][16];
#if HAVE_TDM
  __shared__ _Float16 kbuf[8][2][32 * KROW_];
#endif

  const _Float16* qbase = Qc + ((size_t)s * HW_ + (size_t)qt * 16) * KD_;
  v16h qfrag[4];
#pragma unroll
  for (int ks = 0; ks < 4; ++ks) {
    HPack U;
    const _Float16* qp = qbase + (size_t)row * KD_ + ks * 32 + hi * 16;
    U.h[0] = *(const v8h*)(qp);
    U.h[1] = *(const v8h*)(qp + 8);
    qfrag[ks] = U.v;
  }

  v8f oacc = {};
  float rs[8] = {0.f, 0.f, 0.f, 0.f, 0.f, 0.f, 0.f, 0.f};
  const _Float16* kbase = Kc + (size_t)s * NKTOK_ * KD_;

#if HAVE_TDM
  const int swave = __builtin_amdgcn_readfirstlane(wave);
  const int keyStart = swave * (NKTOK_ / 8);
  const uint32_t kb0 = (uint32_t)(uintptr_t)&kbuf[swave][0][0];
  const uint32_t kb1 = (uint32_t)(uintptr_t)&kbuf[swave][1][0];
  // prologue: fetch first key tile
  tdm_load_2d(kb0, kbase + (size_t)keyStart * KD_, /*f16*/1u,
              /*tile_w=*/KD_, /*tile_h=*/32u, /*row_stride=*/KD_,
              /*pad: 64 DWORD rows, +4 DWORD*/5u, 3u, 1u);
#else
  const int keyStart = wave * (NKTOK_ / 8);
#endif

  for (int kt2 = 0; kt2 < 64; ++kt2) {
    const int key0 = keyStart + kt2 * 32;
#if HAVE_TDM
    if (kt2 + 1 < 64) {   // overlap next tile's DMA with this tile's WMMAs
      tdm_load_2d(((kt2 + 1) & 1) ? kb1 : kb0,
                  kbase + (size_t)(key0 + 32) * KD_, 1u, KD_, 32u, KD_,
                  5u, 3u, 1u);
      __builtin_amdgcn_s_wait_tensorcnt((short)1);
    } else {
      __builtin_amdgcn_s_wait_tensorcnt((short)0);
    }
    const _Float16* ktile = &kbuf[swave][kt2 & 1][0];
#else
    if (kt2 + 1 < 64)
      __builtin_prefetch(kbase + (size_t)(key0 + 32 + row) * KD_, 0, 1);
#endif
#pragma unroll
    for (int half = 0; half < 2; ++half) {
      v8f sacc = {};
#pragma unroll
      for (int ks = 0; ks < 4; ++ks) {
        HPack B;
#if HAVE_TDM
        const _Float16* kp =
            ktile + (half * 16 + row) * KROW_ + ks * 32 + hi * 16;
#else
        const _Float16* kp =
            kbase + (size_t)(key0 + half * 16 + row) * KD_ + ks * 32 + hi * 16;
#endif
        B.h[0] = *(const v8h*)(kp);
        B.h[1] = *(const v8h*)(kp + 8);
        sacc = wmma_f16(qfrag[ks], B.v, sacc);
      }
#pragma unroll
      for (int r = 0; r < 8; ++r) {
        float e = __expf(sacc[r] * 30.0f);
        rs[r] += e;
        pbuf[wave][r + 8 * hi][half * 16 + row] = (_Float16)e;
      }
    }
    HPack PA;
    const _Float16* pp = &pbuf[wave][row][hi * 16];
    PA.h[0] = *(const v8h*)(pp);
    PA.h[1] = *(const v8h*)(pp + 8);
    const int f = key0 >> 10, hw0 = key0 & (HW_ - 1);
    v16h bv = load_cvt16(pos + ((size_t)(f * S_ + s) * CE_ + row) * HW_ +
                         hw0 + hi * 16);
    oacc = wmma_f16(PA.v, bv, oacc);
  }

#pragma unroll
  for (int m = 1; m < 16; m <<= 1)
#pragma unroll
    for (int r = 0; r < 8; ++r) rs[r] += __shfl_xor(rs[r], m, 32);
#pragma unroll
  for (int r = 0; r < 8; ++r) ored[wave][r + 8 * hi][row] = oacc[r];
  if (row == 0) {
#pragma unroll
    for (int r = 0; r < 8; ++r) rsred[wave][r + 8 * hi] = rs[r];
  }
  __syncthreads();

  const int m = tid >> 4, n = tid & 15;
  float o = 0.f, d = 0.f;
#pragma unroll
  for (int w = 0; w < 8; ++w) { o += ored[w][m][n]; d += rsred[w][m]; }
  float val = o / d;
  val = 1.0f / (1.0f + __expf(-val));
  out[((size_t)s * CE_ + n) * HW_ + (size_t)qt * 16 + m] = val;
}

// ---------------------------------------------------------------------------
// Host launch
// ---------------------------------------------------------------------------
extern "C" void kernel_launch(void* const* d_in, const int* in_sizes, int n_in,
                              void* d_out, int out_size, void* d_ws,
                              size_t ws_size, hipStream_t stream) {
  (void)in_sizes; (void)n_in; (void)out_size; (void)ws_size;
  const float* tgt    = (const float*)d_in[0];
  const float* memory = (const float*)d_in[1];
  const float* pos    = (const float*)d_in[2];
  const float* wself  = (const float*)d_in[3];
  const float* bself  = (const float*)d_in[4];
  const float* wcross = (const float*)d_in[5];
  const float* bcross = (const float*)d_in[6];
  float* out = (float*)d_out;

  char* ws = (char*)d_ws;
  _Float16* WhS = (_Float16*)ws;  ws += (size_t)KD_ * C_ * 2;
  _Float16* WhC = (_Float16*)ws;  ws += (size_t)KD_ * C_ * 2;
  _Float16* Qself  = (_Float16*)ws;  ws += (size_t)S_ * HW_ * KD_ * 2;
  _Float16* Qcross = (_Float16*)ws;  ws += (size_t)S_ * HW_ * KD_ * 2;
  _Float16* Kcross = (_Float16*)ws;  ws += (size_t)S_ * NKTOK_ * KD_ * 2;
  float* attn = (float*)ws;          ws += (size_t)S_ * C_ * HW_ * 4;
  float* tn   = (float*)ws;          ws += (size_t)S_ * C_ * HW_ * 4;

  k_convert_w<<<(KD_ * C_ + 255) / 256, 256, 0, stream>>>(wself, wcross, WhS);

  k_proj<<<dim3(HW_ / 16, S_), 256, 0, stream>>>(tgt, 0, WhS, bself, Qself, HW_);
  k_self_attn<<<dim3(HW_ / 16, S_), 256, 0, stream>>>(Qself, tgt, attn);
  k_instnorm<<<S_ * C_, 256, 0, stream>>>(tgt, attn, tn);

  k_proj<<<dim3(HW_ / 16, S_), 256, 0, stream>>>(tn, 0, WhC, bcross, Qcross, HW_);
  k_proj<<<dim3(NKTOK_ / 16, S_), 256, 0, stream>>>(memory, 1, WhC, bcross,
                                                    Kcross, NKTOK_);
  k_cross_attn<<<dim3(HW_ / 16, S_), 256, 0, stream>>>(Qcross, Kcross, pos, out);
}
